// EmbedMatcher_12017318494699
// MI455X (gfx1250) — matrix operand order repaired and checked by reading, not compiled, not run on hardware
//
#include <hip/hip_runtime.h>
#include <hip/hip_bf16.h>
#include <math.h>

// ---------------------------------------------------------------------------
// EmbedMatcher for MI455X (gfx1250, wave32).
//
// Phase 1 (gather-bound, L2-resident): neighbor encoder. Per (row,side):
//   cosine top-32 over 128 struct neighbors + 64 knn neighbors, mean of
//   selected rows (rank-compacted, 4-wide MLP), 256->128 matvec through
//   gcn_W (mean pulled inside the matmul: exact), sigmoid gate fuse.
// Phase 2/3 (WMMA): support-encoder MLP + LayerNorm, then 4 LSTM matcher
//   steps. x@W_ih.T + h_r@W_hh.T is fused into one K=768 GEMM over packed
//   activations [qg|h|r] and packed weights [W_ih|W_hh]. softmax over a
//   length-1 axis == 1, so r = support_g exactly.
// GEMMs: bf16 v_wmma_f32_16x16x32_bf16, f32 accumulate, double-buffered LDS
// with register-staged b128 global loads overlapped with the WMMAs.
// ---------------------------------------------------------------------------

#define V_SYM   200000
#define E_DIM   128
#define PAD_IDX (V_SYM - 1)
#define D_MODEL 256
#define H_DIM   512
#define BQ_N    2048
#define BS_N    5
#define NN_N    128
#define KT_N    64
#define KSEL    32
#define ZCOLS   2048   // 4*H
#define KCAT    768    // D + H
#define STEPS   4

typedef __attribute__((ext_vector_type(16))) __bf16 v16bf;
typedef __attribute__((ext_vector_type(8)))  float  v8f;
typedef __attribute__((ext_vector_type(4)))  float  f4;

union BF4 { __bf16 h[4]; unsigned long long u64; };

__device__ __forceinline__ float sigmf(float x) { return 1.0f / (1.0f + expf(-x)); }

template<int NV>
__device__ __forceinline__ void dot_nsq(const f4* __restrict__ a,
                                        const f4* __restrict__ c,
                                        float& dp, float& nq)
{
  #pragma unroll
  for (int j = 0; j < NV; ++j) {
    f4 v = a[j], w = c[j];
    dp += v[0]*w[0] + v[1]*w[1] + v[2]*w[2] + v[3]*w[3];
    nq += v[0]*v[0] + v[1]*v[1] + v[2]*v[2] + v[3]*v[3];
  }
}

// ===========================================================================
// Phase 1: neighbor encoder. One block (256 threads) per output row.
// (the *_deg inputs are unused by the reference and are ignored here too)
// ===========================================================================
struct __align__(16) NEShared {
  float center[E_DIM];
  float red[512];
  float sim[NN_N];
  float mean[2 * E_DIM];
  float structural[E_DIM];
  float knn_m[E_DIM];
  int   ids[2 * NN_N];   // [0..127] rel ids, [128..255] ent ids
  int   slot[KSEL];      // rank-compacted selected neighbor indices
  int   kidx[KT_N];
  int   eid;
  float cnorm;
  float alpha;
};

__global__ void __launch_bounds__(256)
neighbor_encoder(const float* __restrict__ emb,
                 const float* __restrict__ gcn_W,    // [128,256] row-major
                 const float* __restrict__ gcn_wb,   // [128]
                 const float* __restrict__ gate_W,   // [256]
                 const float* __restrict__ gate_b,   // [1]
                 const int*   __restrict__ knn_tab,  // [V,64]
                 const int*   __restrict__ conn,     // [rows,128,2]
                 const int*   __restrict__ ids2,     // [rows,2]
                 int col,
                 float* __restrict__ out)            // row stride 256 (caller pre-offsets by 0/128)
{
  __shared__ NEShared S;
  const int b = blockIdx.x;
  const int t = threadIdx.x;

  if (t == 0) {
    int e = ids2[b * 2 + col];
    S.eid = (e < 0 || e >= V_SYM) ? PAD_IDX : e;
  }
  __syncthreads();

  // center row + neighbor id cache (int2 = one b64 per thread)
  if (t < E_DIM) {
    S.center[t] = emb[(size_t)S.eid * E_DIM + t];
    int2 pr = ((const int2*)conn)[(size_t)b * NN_N + t];
    S.ids[t]        = (pr.x < 0 || pr.x >= V_SYM) ? PAD_IDX : pr.x;
    S.ids[NN_N + t] = (pr.y < 0 || pr.y >= V_SYM) ? PAD_IDX : pr.y;
  }
  __syncthreads();

  // ||center||
  S.red[t] = (t < E_DIM) ? S.center[t] * S.center[t] : 0.0f;
  __syncthreads();
  for (int s = 128; s > 0; s >>= 1) { if (t < s) S.red[t] += S.red[t + s]; __syncthreads(); }
  if (t == 0) S.cnorm = fmaxf(sqrtf(S.red[0]), 1e-8f);
  __syncthreads();

  // ---- structural branch: cosine sims (2 threads per neighbor, b128 reads) ----
  {
    int n = t & 127, half = t >> 7;
    const f4* erow = (const f4*)(emb + (size_t)S.ids[NN_N + n] * E_DIM + half * 64);
    const f4* crow = (const f4*)(S.center + half * 64);
    float dp = 0.0f, nq = 0.0f;
    dot_nsq<16>(erow, crow, dp, nq);
    S.red[t] = dp; S.red[256 + t] = nq;
  }
  __syncthreads();
  if (t < NN_N) {
    float dot = S.red[t] + S.red[t + 128];
    float nsq = S.red[256 + t] + S.red[256 + t + 128];
    S.sim[t] = dot / (S.cnorm * fmaxf(sqrtf(nsq), 1e-8f));
  }
  __syncthreads();

  // exact top_k tie semantics (lower index wins); rank is unique => it is
  // also the compaction slot for the selected neighbor list.
  if (t < NN_N) {
    float sv = S.sim[t]; int rank = 0;
    for (int j = 0; j < NN_N; ++j) {
      float sj = S.sim[j];
      rank += (sj > sv) || (sj == sv && j < t);
    }
    if (rank < KSEL) S.slot[rank] = t;
  }
  __syncthreads();

  // mean of selected [rel|ent] rows: 32 unconditional gathers, 4-wide MLP
  {
    int part = t >> 7, d = t & 127;
    float acc = 0.0f;
    #pragma unroll
    for (int s = 0; s < KSEL; s += 4) {
      float a0 = emb[(size_t)S.ids[part * NN_N + S.slot[s + 0]] * E_DIM + d];
      float a1 = emb[(size_t)S.ids[part * NN_N + S.slot[s + 1]] * E_DIM + d];
      float a2 = emb[(size_t)S.ids[part * NN_N + S.slot[s + 2]] * E_DIM + d];
      float a3 = emb[(size_t)S.ids[part * NN_N + S.slot[s + 3]] * E_DIM + d];
      acc += (a0 + a1) + (a2 + a3);
    }
    S.mean[t] = acc * (1.0f / KSEL);
  }
  __syncthreads();

  // structural = tanh(gcn_W @ mean + wb)  (2 threads per output, b128 reads)
  {
    int j = t & 127, half = t >> 7;
    const f4* wr = (const f4*)(gcn_W + (size_t)j * 256 + half * 128);
    const f4* mr = (const f4*)(S.mean + half * 128);
    float dp = 0.0f;
    #pragma unroll
    for (int d = 0; d < 32; ++d) {
      f4 a = wr[d], m = mr[d];
      dp += a[0]*m[0] + a[1]*m[1] + a[2]*m[2] + a[3]*m[3];
    }
    S.red[t] = dp;
  }
  __syncthreads();
  if (t < E_DIM) S.structural[t] = tanhf(S.red[t] + S.red[t + 128] + gcn_wb[t]);
  __syncthreads();

  // ---- KNN branch ----
  if (t < KT_N) {
    int k = knn_tab[(size_t)S.eid * KT_N + t];
    S.kidx[t] = (k < 0 || k >= V_SYM) ? PAD_IDX : k;
  }
  __syncthreads();
  {
    int n = t & 63, q = t >> 6;
    const f4* erow = (const f4*)(emb + (size_t)S.kidx[n] * E_DIM + q * 32);
    const f4* crow = (const f4*)(S.center + q * 32);
    float dp = 0.0f, nq = 0.0f;
    dot_nsq<8>(erow, crow, dp, nq);
    S.red[t] = dp; S.red[256 + t] = nq;
  }
  __syncthreads();
  if (t < KT_N) {
    float dot = S.red[t] + S.red[64 + t] + S.red[128 + t] + S.red[192 + t];
    float nsq = S.red[256 + t] + S.red[320 + t] + S.red[384 + t] + S.red[448 + t];
    S.sim[t] = dot / (S.cnorm * fmaxf(sqrtf(nsq), 1e-8f));
  }
  __syncthreads();
  if (t < KT_N) {
    float sv = S.sim[t]; int rank = 0;
    for (int j = 0; j < KT_N; ++j) {
      float sj = S.sim[j];
      rank += (sj > sv) || (sj == sv && j < t);
    }
    if (rank < KSEL) S.slot[rank] = t;
  }
  __syncthreads();
  // mean2: rel half is 32 copies of emb[PAD] -> just emb[PAD]; ent half is
  // the mean of the 32 selected knn rows (4-wide MLP gathers)
  {
    int part = t >> 7, d = t & 127;
    if (part == 0) {
      S.mean[t] = emb[(size_t)PAD_IDX * E_DIM + d];
    } else {
      float acc = 0.0f;
      #pragma unroll
      for (int s = 0; s < KSEL; s += 4) {
        float a0 = emb[(size_t)S.kidx[S.slot[s + 0]] * E_DIM + d];
        float a1 = emb[(size_t)S.kidx[S.slot[s + 1]] * E_DIM + d];
        float a2 = emb[(size_t)S.kidx[S.slot[s + 2]] * E_DIM + d];
        float a3 = emb[(size_t)S.kidx[S.slot[s + 3]] * E_DIM + d];
        acc += (a0 + a1) + (a2 + a3);
      }
      S.mean[t] = acc * (1.0f / KSEL);
    }
  }
  __syncthreads();
  {
    int j = t & 127, half = t >> 7;
    const f4* wr = (const f4*)(gcn_W + (size_t)j * 256 + half * 128);
    const f4* mr = (const f4*)(S.mean + half * 128);
    float dp = 0.0f;
    #pragma unroll
    for (int d = 0; d < 32; ++d) {
      f4 a = wr[d], m = mr[d];
      dp += a[0]*m[0] + a[1]*m[1] + a[2]*m[2] + a[3]*m[3];
    }
    S.red[t] = dp;
  }
  __syncthreads();
  if (t < E_DIM) S.knn_m[t] = tanhf(S.red[t] + S.red[t + 128] + gcn_wb[t]);
  __syncthreads();

  // sigmoid gate over concat(structural, knn_mean)
  S.red[t] = (t < E_DIM) ? S.structural[t] * gate_W[t] : S.knn_m[t - 128] * gate_W[t];
  __syncthreads();
  for (int s = 128; s > 0; s >>= 1) { if (t < s) S.red[t] += S.red[t + s]; __syncthreads(); }
  if (t == 0) S.alpha = sigmf(S.red[0] + gate_b[0]);
  __syncthreads();

  if (t < E_DIM)
    out[(size_t)b * D_MODEL + t] =
        (1.0f - S.alpha) * S.structural[t] + S.alpha * S.knn_m[t];
}

// ===========================================================================
// WMMA bf16 GEMM:  C[M,N] = A[M,K] @ W[N,K]^T (+bias)(+relu | +residual)
// 256 threads = 8 waves; block tile 32x128; K-chunk 32; per wave: one A frag,
// two B frags, two v_wmma_f32_16x16x32_bf16 per chunk. Double-buffered LDS:
// next tile's 5 b128 global loads sit in registers while the WMMAs run on
// the current buffer; one barrier per K iteration.
// ===========================================================================
#define GBM 32
#define GBN 128
#define GBK 32
#define LDP (GBK + 8)   // bf16 row stride: 40 elems = 80B (8B-aligned groups)

__global__ void __launch_bounds__(256)
wmma_gemm(const float* __restrict__ A, const float* __restrict__ W,
          const float* __restrict__ bias, const float* __restrict__ resid,
          float* __restrict__ C,
          int M, int N, int K, int lda, int ldw, int ldc, int mode)
{
  __shared__ __align__(16) __bf16 As[2][GBM][LDP];
  __shared__ __align__(16) __bf16 Ws[2][GBN][LDP];

  const int tid  = threadIdx.x;
  const int bm0  = blockIdx.y * GBM;
  const int bn0  = blockIdx.x * GBN;
  const int wave = tid >> 5, lane = tid & 31;
  const int wm   = wave & 1, wn = wave >> 1;       // waves: 2 (M) x 4 (N)
  const int r    = lane & 15, hi = lane >> 4;

  // per-thread staging coordinates (one float4 per tile slice)
  const int am  = tid >> 3;            // 0..31
  const int ak0 = (tid & 7) * 4;       // 0,4,..,28
  const int arow = bm0 + am;
  const size_t aoff = (size_t)((arow < M) ? arow : (M > 0 ? M - 1 : 0)) * lda + ak0;
  const float amsk = (arow < M) ? 1.0f : 0.0f;

  size_t woff[4]; float wmsk[4]; int wn_[4], wk_[4];
  #pragma unroll
  for (int i = 0; i < 4; ++i) {
    int e4 = tid + i * 256;
    int n  = e4 >> 3, k0 = (e4 & 7) * 4;
    int row = bn0 + n;
    int rc  = (row < N) ? row : (N - 1);
    woff[i] = (size_t)rc * ldw + k0;
    wmsk[i] = (row < N) ? 1.0f : 0.0f;
    wn_[i] = n; wk_[i] = k0;
  }

  v8f acc0 = {}; v8f acc1 = {};
  const int nk = K / GBK;

  f4 va, vw0, vw1, vw2, vw3;
  // prologue: tile 0 -> regs -> LDS[0]
  va  = *(const f4*)(A + aoff);
  vw0 = *(const f4*)(W + woff[0]);
  vw1 = *(const f4*)(W + woff[1]);
  vw2 = *(const f4*)(W + woff[2]);
  vw3 = *(const f4*)(W + woff[3]);
  {
    BF4 pk;
    pk.h[0] = (__bf16)(va[0] * amsk); pk.h[1] = (__bf16)(va[1] * amsk);
    pk.h[2] = (__bf16)(va[2] * amsk); pk.h[3] = (__bf16)(va[3] * amsk);
    *(unsigned long long*)(&As[0][am][ak0]) = pk.u64;
    f4 vv[4] = { vw0, vw1, vw2, vw3 };
    #pragma unroll
    for (int i = 0; i < 4; ++i) {
      BF4 qk;
      qk.h[0] = (__bf16)(vv[i][0] * wmsk[i]); qk.h[1] = (__bf16)(vv[i][1] * wmsk[i]);
      qk.h[2] = (__bf16)(vv[i][2] * wmsk[i]); qk.h[3] = (__bf16)(vv[i][3] * wmsk[i]);
      *(unsigned long long*)(&Ws[0][wn_[i]][wk_[i]]) = qk.u64;
    }
  }
  __syncthreads();

  for (int it = 0; it < nk; ++it) {
    const int cur  = it & 1;
    const bool more = (it + 1 < nk);
    const int kn = (it + 1) * GBK;

    // issue next tile's global loads (in flight during the WMMAs below)
    if (more) {
      va  = *(const f4*)(A + aoff + kn);
      vw0 = *(const f4*)(W + woff[0] + kn);
      vw1 = *(const f4*)(W + woff[1] + kn);
      vw2 = *(const f4*)(W + woff[2] + kn);
      vw3 = *(const f4*)(W + woff[3] + kn);
    }

    // Build fragments per CDNA5 ISA layouts (CDNA5 ISA):
    //  A 16x32 bf16: lane r (r<16) VGPR v pair p -> K = 2*(v&3)+p + 8*hi + 16*(v>=4)
    //  B 32x16 bf16: element e holds K = (e&7) + 8*hi + 16*(e>=8) (same as A)
    v16bf af, fb0, fb1;
    #pragma unroll
    for (int e = 0; e < 16; ++e) {
      int v = e >> 1, p = e & 1;
      int ka = ((v & 3) * 2 + p) + hi * 8 + ((v & 4) ? 16 : 0);
      af[e] = As[cur][wm * 16 + r][ka];
      int kb = (e & 7) + hi * 8 + ((e & 8) ? 16 : 0);
      fb0[e] = Ws[cur][wn * 32 + r][kb];
      fb1[e] = Ws[cur][wn * 32 + 16 + r][kb];
    }
    acc0 = __builtin_amdgcn_wmma_f32_16x16x32_bf16(false, af, false, fb0,
                                                   (short)0, acc0, false, false);
    acc1 = __builtin_amdgcn_wmma_f32_16x16x32_bf16(false, af, false, fb1,
                                                   (short)0, acc1, false, false);

    if (more) {
      const int nxt = cur ^ 1;
      BF4 pk;
      pk.h[0] = (__bf16)(va[0] * amsk); pk.h[1] = (__bf16)(va[1] * amsk);
      pk.h[2] = (__bf16)(va[2] * amsk); pk.h[3] = (__bf16)(va[3] * amsk);
      *(unsigned long long*)(&As[nxt][am][ak0]) = pk.u64;
      f4 vv[4] = { vw0, vw1, vw2, vw3 };
      #pragma unroll
      for (int i = 0; i < 4; ++i) {
        BF4 qk;
        qk.h[0] = (__bf16)(vv[i][0] * wmsk[i]); qk.h[1] = (__bf16)(vv[i][1] * wmsk[i]);
        qk.h[2] = (__bf16)(vv[i][2] * wmsk[i]); qk.h[3] = (__bf16)(vv[i][3] * wmsk[i]);
        *(unsigned long long*)(&Ws[nxt][wn_[i]][wk_[i]]) = qk.u64;
      }
      __syncthreads();
    }
  }

  // C/D layout: VGPR i, lanes 0-15 -> M=i, lanes 16-31 -> M=i+8; N = lane&15
  #pragma unroll
  for (int i = 0; i < 8; ++i) {
    int row = bm0 + wm * 16 + i + hi * 8;
    #pragma unroll
    for (int s2 = 0; s2 < 2; ++s2) {
      int colg = bn0 + wn * 32 + s2 * 16 + r;
      if (row < M && colg < N) {
        float v = (s2 == 0) ? acc0[i] : acc1[i];
        if (bias) v += bias[colg];
        if (mode == 1)      v = fmaxf(v, 0.0f);
        else if (mode == 2) v += resid[(size_t)row * ldc + colg];
        C[(size_t)row * ldc + colg] = v;
      }
    }
  }
}

// ===========================================================================
// Small elementwise / reduction kernels
// ===========================================================================
__global__ void __launch_bounds__(256)
layernorm_rows(const float* __restrict__ in, const float* __restrict__ g,
               const float* __restrict__ b, float* __restrict__ out)
{
  __shared__ float red[256];
  int row = blockIdx.x, t = threadIdx.x;
  float x = in[(size_t)row * D_MODEL + t];
  red[t] = x; __syncthreads();
  for (int s = 128; s > 0; s >>= 1) { if (t < s) red[t] += red[t + s]; __syncthreads(); }
  float mu = red[0] * (1.0f / D_MODEL);
  __syncthreads();
  float d = x - mu;
  red[t] = d * d; __syncthreads();
  for (int s = 128; s > 0; s >>= 1) { if (t < s) red[t] += red[t + s]; __syncthreads(); }
  float var = red[0] * (1.0f / D_MODEL);
  out[(size_t)row * D_MODEL + t] = d * rsqrtf(var + 1e-5f) * g[t] + b[t];
}

__global__ void __launch_bounds__(256)
mean_rows5(const float* __restrict__ in, float* __restrict__ out)
{
  int t = threadIdx.x;
  float s = 0.0f;
  for (int r = 0; r < BS_N; ++r) s += in[(size_t)r * D_MODEL + t];
  out[t] = s * (1.0f / BS_N);
}

__global__ void __launch_bounds__(256)
pack_wcat(const float* __restrict__ W_ih, const float* __restrict__ W_hh,
          const float* __restrict__ b_ih, const float* __restrict__ b_hh,
          float* __restrict__ W_cat, float* __restrict__ bias_cat)
{
  int n = blockIdx.x;  // 0..2047
  for (int k = threadIdx.x; k < KCAT; k += 256)
    W_cat[(size_t)n * KCAT + k] =
        (k < D_MODEL) ? W_ih[(size_t)n * D_MODEL + k]
                      : W_hh[(size_t)n * H_DIM + (k - D_MODEL)];
  if (threadIdx.x == 0) bias_cat[n] = b_ih[n] + b_hh[n];
}

__global__ void __launch_bounds__(256)
init_state(const float* __restrict__ qg, float* __restrict__ a_cat,
           float* __restrict__ c_state)
{
  int b = blockIdx.x;
  for (int k = threadIdx.x; k < KCAT; k += 256)
    a_cat[(size_t)b * KCAT + k] = (k < D_MODEL) ? qg[(size_t)b * D_MODEL + k] : 0.0f;
  for (int d = threadIdx.x; d < H_DIM; d += 256)
    c_state[(size_t)b * H_DIM + d] = 0.0f;
}

// softmax over the length-1 support axis is identically 1 => r = support_g
__global__ void __launch_bounds__(256)
lstm_pointwise(const float* __restrict__ z, const float* __restrict__ qg,
               const float* __restrict__ sg, float* __restrict__ c_state,
               float* __restrict__ a_cat, float* __restrict__ qf, int last)
{
  int b = blockIdx.x;
  const float* zr = z + (size_t)b * ZCOLS;
  for (int d = threadIdx.x; d < H_DIM; d += 256) {
    float iv = zr[d], fv = zr[H_DIM + d], gv = zr[2 * H_DIM + d], ov = zr[3 * H_DIM + d];
    float cprev = c_state[(size_t)b * H_DIM + d];
    float cn = sigmf(fv) * cprev + sigmf(iv) * tanhf(gv);
    c_state[(size_t)b * H_DIM + d] = cn;
    float h2 = sigmf(ov) * tanhf(cn);
    if (d < D_MODEL) {
      float h = qg[(size_t)b * D_MODEL + d] + h2;     // h = query + h2[:, :D]
      a_cat[(size_t)b * KCAT + D_MODEL + d] = h;      // h part of h_r
      a_cat[(size_t)b * KCAT + 2 * D_MODEL + d] = sg[d];  // r part (attn == 1)
      if (last) qf[(size_t)b * D_MODEL + d] = h;
    }
  }
}

__global__ void __launch_bounds__(256)
normalize_vec(const float* __restrict__ in, float* __restrict__ out)
{
  __shared__ float red[256];
  int t = threadIdx.x;
  float x = in[t];
  red[t] = x * x; __syncthreads();
  for (int s = 128; s > 0; s >>= 1) { if (t < s) red[t] += red[t + s]; __syncthreads(); }
  out[t] = x / fmaxf(sqrtf(red[0]), 1e-12f);
}

__global__ void __launch_bounds__(256)
final_cosine(const float* __restrict__ qf, const float* __restrict__ sg_hat,
             float* __restrict__ out)
{
  __shared__ float red[512];
  int b = blockIdx.x, t = threadIdx.x;
  float x = qf[(size_t)b * D_MODEL + t];
  red[t] = x * x;
  red[256 + t] = x * sg_hat[t];
  __syncthreads();
  for (int s = 128; s > 0; s >>= 1) {
    if (t < s) { red[t] += red[t + s]; red[256 + t] += red[256 + t + s]; }
    __syncthreads();
  }
  if (t == 0) out[b] = red[256] / fmaxf(sqrtf(red[0]), 1e-12f);
}

// ===========================================================================
extern "C" void kernel_launch(void* const* d_in, const int* in_sizes, int n_in,
                              void* d_out, int out_size, void* d_ws, size_t ws_size,
                              hipStream_t stream)
{
  const float* emb    = (const float*)d_in[0];
  const float* gcn_W  = (const float*)d_in[1];
  const float* gcn_wb = (const float*)d_in[2];
  const float* gate_W = (const float*)d_in[3];
  const float* gate_b = (const float*)d_in[4];
  const float* se_w1  = (const float*)d_in[5];
  const float* se_b1  = (const float*)d_in[6];
  const float* se_w2  = (const float*)d_in[7];
  const float* se_b2  = (const float*)d_in[8];
  const float* ln_g   = (const float*)d_in[9];
  const float* ln_b   = (const float*)d_in[10];
  const float* W_ih   = (const float*)d_in[11];
  const float* W_hh   = (const float*)d_in[12];
  const float* b_ih   = (const float*)d_in[13];
  const float* b_hh   = (const float*)d_in[14];
  const int* query    = (const int*)d_in[15];
  const int* support  = (const int*)d_in[16];
  const int* q_l_conn = (const int*)d_in[17];
  const int* q_r_conn = (const int*)d_in[19];
  const int* s_l_conn = (const int*)d_in[21];
  const int* s_r_conn = (const int*)d_in[23];
  const int* knn_tab  = (const int*)d_in[25];
  (void)in_sizes; (void)n_in; (void)out_size; (void)ws_size;

  // workspace carve-up (~46 MB of f32)
  float* ws = (float*)d_ws;
  size_t off = 0;
  auto alloc = [&](size_t n) { float* p = ws + off; off += n; return p; };
  float* query_n   = alloc((size_t)BQ_N * D_MODEL);
  float* supp_n    = alloc((size_t)BS_N * D_MODEL);
  float* t1q       = alloc((size_t)BQ_N * H_DIM);
  float* hq        = alloc((size_t)BQ_N * D_MODEL);
  float* query_g   = alloc((size_t)BQ_N * D_MODEL);
  float* t1s       = alloc((size_t)BS_N * H_DIM);
  float* hs        = alloc((size_t)BS_N * D_MODEL);
  float* se_s      = alloc((size_t)BS_N * D_MODEL);
  float* support_g = alloc(D_MODEL);
  float* sg_hat    = alloc(D_MODEL);
  float* W_cat     = alloc((size_t)ZCOLS * KCAT);
  float* bias_cat  = alloc(ZCOLS);
  float* A_cat     = alloc((size_t)BQ_N * KCAT);
  float* z_buf     = alloc((size_t)BQ_N * ZCOLS);
  float* c_state   = alloc((size_t)BQ_N * H_DIM);
  float* qf        = alloc((size_t)BQ_N * D_MODEL);

  // ---- Phase 1: neighbor encoders (left/right halves of each row) ----
  neighbor_encoder<<<BQ_N, 256, 0, stream>>>(emb, gcn_W, gcn_wb, gate_W, gate_b,
                                             knn_tab, q_l_conn, query, 0, query_n);
  neighbor_encoder<<<BQ_N, 256, 0, stream>>>(emb, gcn_W, gcn_wb, gate_W, gate_b,
                                             knn_tab, q_r_conn, query, 1, query_n + E_DIM);
  neighbor_encoder<<<BS_N, 256, 0, stream>>>(emb, gcn_W, gcn_wb, gate_W, gate_b,
                                             knn_tab, s_l_conn, support, 0, supp_n);
  neighbor_encoder<<<BS_N, 256, 0, stream>>>(emb, gcn_W, gcn_wb, gate_W, gate_b,
                                             knn_tab, s_r_conn, support, 1, supp_n + E_DIM);

  // ---- Phase 2: support encoder (MLP + residual + LayerNorm) ----
  wmma_gemm<<<dim3(H_DIM / GBN, BQ_N / GBM), 256, 0, stream>>>(
      query_n, se_w1, se_b1, nullptr, t1q,
      BQ_N, H_DIM, D_MODEL, D_MODEL, D_MODEL, H_DIM, /*relu*/1);
  wmma_gemm<<<dim3(D_MODEL / GBN, BQ_N / GBM), 256, 0, stream>>>(
      t1q, se_w2, se_b2, query_n, hq,
      BQ_N, D_MODEL, H_DIM, H_DIM, H_DIM, D_MODEL, /*resid*/2);
  layernorm_rows<<<BQ_N, 256, 0, stream>>>(hq, ln_g, ln_b, query_g);

  // support path (M=5, row-guarded via clamp+mask)
  wmma_gemm<<<dim3(H_DIM / GBN, 1), 256, 0, stream>>>(
      supp_n, se_w1, se_b1, nullptr, t1s,
      BS_N, H_DIM, D_MODEL, D_MODEL, D_MODEL, H_DIM, 1);
  wmma_gemm<<<dim3(D_MODEL / GBN, 1), 256, 0, stream>>>(
      t1s, se_w2, se_b2, supp_n, hs,
      BS_N, D_MODEL, H_DIM, H_DIM, H_DIM, D_MODEL, 2);
  layernorm_rows<<<BS_N, 256, 0, stream>>>(hs, ln_g, ln_b, se_s);
  mean_rows5<<<1, 256, 0, stream>>>(se_s, support_g);

  // ---- Phase 3: LSTM matching network (4 steps, fused K=768 GEMM) ----
  pack_wcat<<<ZCOLS, 256, 0, stream>>>(W_ih, W_hh, b_ih, b_hh, W_cat, bias_cat);
  init_state<<<BQ_N, 256, 0, stream>>>(query_g, A_cat, c_state);
  for (int step = 0; step < STEPS; ++step) {
    wmma_gemm<<<dim3(ZCOLS / GBN, BQ_N / GBM), 256, 0, stream>>>(
        A_cat, W_cat, bias_cat, nullptr, z_buf,
        BQ_N, ZCOLS, KCAT, KCAT, KCAT, ZCOLS, 0);
    lstm_pointwise<<<BQ_N, 256, 0, stream>>>(z_buf, query_g, support_g,
                                             c_state, A_cat, qf,
                                             step == STEPS - 1);
  }

  // ---- Final: cosine(query_f, support_g) ----
  normalize_vec<<<1, 256, 0, stream>>>(support_g, sg_hat);
  final_cosine<<<BQ_N, 256, 0, stream>>>(qf, sg_hat, (float*)d_out);
}
